// Encoder_55370718380126
// MI455X (gfx1250) — matrix operand order repaired and checked by reading, not compiled, not run on hardware
//
#include <hip/hip_runtime.h>
#include <hip/hip_bf16.h>
#include <cstdint>
#include <cstddef>

#define DIM   1024
#define HEADS 8
#define HD    128
#define NSEQ  1025
#define NTOK  4100   // 4 * 1025
#define NPAD  1056   // 33 * 32, covers all 32-key chunk reads
#define QSCALE 0.03125f  // 1024^-0.5
#define EPSF  1e-5f
#define STR   40     // LDS row stride (bf16 elems), 80B = 16B-aligned
#define BM    128
#define BN    128

typedef __attribute__((ext_vector_type(16))) __bf16 v16bf;
typedef __attribute__((ext_vector_type(8)))  __bf16 v8bf;
typedef __attribute__((ext_vector_type(2)))  __bf16 v2bf;
typedef __attribute__((ext_vector_type(8)))  float  v8f;
typedef __attribute__((ext_vector_type(4)))  float  v4f;
typedef __attribute__((ext_vector_type(2)))  unsigned int v2u;

__device__ __forceinline__ v8f zero_v8f() {
  v8f z;
  #pragma unroll
  for (int i = 0; i < 8; ++i) z[i] = 0.f;
  return z;
}

__device__ __forceinline__ v8bf zero_v8bf() {
  v8bf z;
  #pragma unroll
  for (int i = 0; i < 8; ++i) z[i] = (__bf16)0.f;
  return z;
}

// pack two floats -> one dword of two bf16 (lo at low half); should lower to
// a single v_cvt_pk_bf16_f32
__device__ __forceinline__ unsigned int pkbf(float lo, float hi) {
  v2bf p;
  p.x = (__bf16)lo;
  p.y = (__bf16)hi;
  union { v2bf v; unsigned int u; } cv;
  cv.v = p;
  return cv.u;
}

// Build a 16-element bf16 fragment from two 16-byte chunks.
// A-frag (16x32, MxK): lane half h: chunks at K = 8h and 8h+16  -> p1 = p0+16
// B-frag (32x16, KxN): lane half h: chunks at K = 8h and 8h+16  -> p1 = p0+16
__device__ __forceinline__ v16bf mk_frag(const __bf16* p0, const __bf16* p1) {
  union { v16bf v; v8bf h[2]; } u;
  u.h[0] = *(const v8bf*)p0;
  u.h[1] = *(const v8bf*)p1;
  return u.v;
}

__device__ __forceinline__ v8f wmma_bf16(v16bf a, v16bf b, v8f c) {
  return __builtin_amdgcn_wmma_f32_16x16x32_bf16(false, a, false, b, (short)0, c, false, false);
}

// ---------------------------------------------------------------------------
// GEMM: C[M,Nc] = A_bf16[M,K] @ W_f32->bf16[K,Nc], fused epilogues.
// 128x128 tile, 8 waves (each 32x64 = 2x4 WMMA tiles), double-buffered LDS.
// mode 0: split qkv -> q(*QSCALE)/k [B,H,NPAD,HD], v [B,H,HD,NPAD] (bf16)
// mode 1: outf = resid + acc + bias   (fp32, in-place residual OK)
// mode 2: outh = gelu(acc + bias)     (bf16)
// ---------------------------------------------------------------------------
__global__ __launch_bounds__(256) void gemm_kernel(
    const __bf16* __restrict__ A, const float* __restrict__ W,
    const float* __restrict__ bias, const float* __restrict__ resid,
    float* __restrict__ outf, __bf16* __restrict__ outh,
    __bf16* __restrict__ qb, __bf16* __restrict__ kb, __bf16* __restrict__ vb,
    int M, int K, int Nc, int mode)
{
  __shared__ __bf16 As[2][BM * STR];
  __shared__ __bf16 Bs[2][BN * STR];

  const int t    = threadIdx.x;
  const int lane = t & 31;
  const int wave = t >> 5;
  const int half = lane >> 4;
  const int ln   = lane & 15;

  const int mrow0 = blockIdx.x * BM;
  const int ncol0 = blockIdx.y * BN;
  const int wm = (wave >> 1) * 32;   // 0,32,64,96
  const int wn = (wave & 1) * 64;    // 0,64

  // A staging: 128 rows x 32 cols bf16; thread -> row t>>1, 16 elems
  const int ar = t >> 1;
  const int ac = (t & 1) * 16;
  const int agm = mrow0 + ar;
  // B staging: 32 k-rows x 128 cols fp32; thread owns 4 consecutive k rows
  const int kq = (t >> 5) * 4;       // 0,4,...,28
  const int bn = (t & 31) * 4;       // 0..124

  const int nk = K >> 5;

  v8f acc[2][4];
  #pragma unroll
  for (int mt = 0; mt < 2; ++mt)
    #pragma unroll
    for (int nt = 0; nt < 4; ++nt) acc[mt][nt] = zero_v8f();

  v8bf a_st0, a_st1;
  v4f  b_st[4];

  auto loadg = [&](int kc) {
    a_st0 = zero_v8bf(); a_st1 = zero_v8bf();
    if (agm < M) {
      const __bf16* ap = A + (size_t)agm * K + kc + ac;
      a_st0 = *(const v8bf*)ap;
      a_st1 = *(const v8bf*)(ap + 8);
    }
    #pragma unroll
    for (int i = 0; i < 4; ++i)
      b_st[i] = *(const v4f*)(W + (size_t)(kc + kq + i) * Nc + ncol0 + bn);
  };
  auto store_lds = [&](int p) {
    *(v8bf*)&As[p][ar * STR + ac]     = a_st0;
    *(v8bf*)&As[p][ar * STR + ac + 8] = a_st1;
    #pragma unroll
    for (int j = 0; j < 4; ++j) {
      v2u u;
      u.x = pkbf(b_st[0][j], b_st[1][j]);
      u.y = pkbf(b_st[2][j], b_st[3][j]);
      *(v2u*)&Bs[p][(bn + j) * STR + kq] = u;   // 8B store, 4 consecutive k
    }
  };

  loadg(0);
  store_lds(0);
  __syncthreads();

  for (int i = 0; i < nk; ++i) {
    const int p = i & 1;
    const bool more = (i + 1 < nk);
    if (more) loadg((i + 1) * 32);

    v16bf af[2], bfr[4];
    #pragma unroll
    for (int mt = 0; mt < 2; ++mt) {
      const __bf16* ap = &As[p][(wm + mt * 16 + ln) * STR + half * 8];
      af[mt] = mk_frag(ap, ap + 16);
    }
    #pragma unroll
    for (int nt = 0; nt < 4; ++nt) {
      const __bf16* bp = &Bs[p][(wn + nt * 16 + ln) * STR + half * 8];
      bfr[nt] = mk_frag(bp, bp + 16);
    }
    #pragma unroll
    for (int mt = 0; mt < 2; ++mt)
      #pragma unroll
      for (int nt = 0; nt < 4; ++nt)
        acc[mt][nt] = wmma_bf16(af[mt], bfr[nt], acc[mt][nt]);

    if (more) store_lds(1 - p);
    __syncthreads();
  }

  #pragma unroll
  for (int mt = 0; mt < 2; ++mt) {
    #pragma unroll
    for (int nt = 0; nt < 4; ++nt) {
      v8f a = acc[mt][nt];
      const int col  = ncol0 + wn + nt * 16 + ln;
      const int row0 = mrow0 + wm + mt * 16 + half * 8;
      #pragma unroll
      for (int i = 0; i < 8; ++i) {
        int row = row0 + i;
        if (row >= M) continue;
        float v = a[i];
        if (mode == 0) {
          int b = row / NSEQ, n = row % NSEQ;
          int which = col >> 10, h = (col & 1023) >> 7, d = col & 127;
          size_t bh = (size_t)(b * HEADS + h);
          if (which == 0)      qb[(bh * NPAD + n) * HD + d] = (__bf16)(v * QSCALE);
          else if (which == 1) kb[(bh * NPAD + n) * HD + d] = (__bf16)v;
          else                 vb[(bh * HD + d) * NPAD + n] = (__bf16)v;
        } else if (mode == 1) {
          size_t o = (size_t)row * Nc + col;
          outf[o] = resid[o] + v + bias[col];
        } else {
          float u = v + bias[col];
          float g = 0.5f * u * (1.0f + erff(u * 0.70710678118654752f));
          outh[(size_t)row * Nc + col] = (__bf16)g;
        }
      }
    }
  }
}

// ---------------------------------------------------------------------------
// Flash attention: 4 waves / block, each wave owns one 16-row q-tile.
// Causal mask (src_mask all-true). Online softmax. Wave-private LDS bounce
// for P (C-frag -> A-frag), ordered with s_wait_dscnt (DS in-order per wave).
// ---------------------------------------------------------------------------
__global__ __launch_bounds__(128) void attn_kernel(
    const __bf16* __restrict__ qb, const __bf16* __restrict__ kb,
    const __bf16* __restrict__ vb, __bf16* __restrict__ aout)
{
  __shared__ __bf16 Pl[4][16 * 32];

  const int wave = threadIdx.x >> 5;
  const int lane = threadIdx.x & 31;
  const int qt = blockIdx.x * 4 + wave;   // q-tile index, 0..64 valid
  if (qt > 64) return;
  const int h  = blockIdx.y;
  const int b  = blockIdx.z;
  const int half = lane >> 4;
  const int ln   = lane & 15;

  const size_t bh = (size_t)(b * HEADS + h);
  const __bf16* Q  = qb + bh * NPAD * HD;
  const __bf16* Kp = kb + bh * NPAD * HD;
  const __bf16* Vp = vb + bh * HD * NPAD;
  __bf16* Pw = &Pl[wave][0];

  // Q fragments: 4 K-chunks of 32 along head-dim
  const int qrow = qt * 16 + ln;
  v16bf qf[4];
  #pragma unroll
  for (int kc = 0; kc < 4; ++kc) {
    const __bf16* p = Q + (size_t)qrow * HD + kc * 32 + half * 8;
    qf[kc] = mk_frag(p, p + 16);
  }

  v8f oacc[8];
  #pragma unroll
  for (int d = 0; d < 8; ++d) oacc[d] = zero_v8f();
  float mr[8], lr[8];
  #pragma unroll
  for (int i = 0; i < 8; ++i) { mr[i] = -__builtin_inff(); lr[i] = 0.f; }

  int qmax = qt * 16 + 15; if (qmax > 1024) qmax = 1024;
  const int nch = qmax / 32 + 1;

  for (int c = 0; c < nch; ++c) {
    const int kb0 = c * 32;
    // S = Q @ K^T  (16 x 32 as two 16x16 tiles)
    v8f s0 = zero_v8f(), s1 = zero_v8f();
    #pragma unroll
    for (int kc = 0; kc < 4; ++kc) {
      const __bf16* p0 = Kp + (size_t)(kb0 + ln) * HD + kc * 32 + half * 8;
      const __bf16* p1 = Kp + (size_t)(kb0 + 16 + ln) * HD + kc * 32 + half * 8;
      s0 = wmma_bf16(qf[kc], mk_frag(p0, p0 + 16), s0);
      s1 = wmma_bf16(qf[kc], mk_frag(p1, p1 + 16), s1);
    }
    // mask + online softmax
    float p0v[8], p1v[8], scl[8];
    #pragma unroll
    for (int i = 0; i < 8; ++i) {
      const int qg  = qt * 16 + half * 8 + i;
      const int k0g = kb0 + ln;
      const int k1g = kb0 + 16 + ln;
      const bool rowok = (qg <= 1024);
      float e0 = (rowok && k0g <= qg) ? s0[i] : -__builtin_inff();
      float e1 = (rowok && k1g <= qg) ? s1[i] : -__builtin_inff();
      float rm = fmaxf(e0, e1);
      #pragma unroll
      for (int m = 1; m < 16; m <<= 1) rm = fmaxf(rm, __shfl_xor(rm, m, 32));
      float mn = fmaxf(mr[i], rm);
      float sc = (mr[i] == -__builtin_inff()) ? 0.f : __expf(mr[i] - mn);
      float q0 = (e0 == -__builtin_inff()) ? 0.f : __expf(e0 - mn);
      float q1 = (e1 == -__builtin_inff()) ? 0.f : __expf(e1 - mn);
      float rs = q0 + q1;
      #pragma unroll
      for (int m = 1; m < 16; m <<= 1) rs += __shfl_xor(rs, m, 32);
      lr[i] = lr[i] * sc + rs;
      mr[i] = mn;
      scl[i] = sc; p0v[i] = q0; p1v[i] = q1;
    }
    #pragma unroll
    for (int d = 0; d < 8; ++d)
      #pragma unroll
      for (int i = 0; i < 8; ++i) oacc[d][i] *= scl[i];

    // wave-private P bounce through LDS (C-frag layout -> A-frag layout).
    // DS ops are in-order within a wave; fence keeps the compiler honest.
    asm volatile("s_wait_dscnt 0" ::: "memory");
    #pragma unroll
    for (int i = 0; i < 8; ++i) {
      Pw[(half * 8 + i) * 32 + ln]      = (__bf16)p0v[i];
      Pw[(half * 8 + i) * 32 + 16 + ln] = (__bf16)p1v[i];
    }
    asm volatile("s_wait_dscnt 0" ::: "memory");
    const __bf16* pp = &Pw[ln * 32 + half * 8];
    v16bf pf = mk_frag(pp, pp + 16);
    // O += P @ V   (V stored [dim][key] -> contiguous B-frag loads)
    #pragma unroll
    for (int d = 0; d < 8; ++d) {
      const __bf16* vp = Vp + (size_t)(d * 16 + ln) * NPAD + kb0 + half * 8;
      oacc[d] = wmma_bf16(pf, mk_frag(vp, vp + 16), oacc[d]);
    }
  }

  #pragma unroll
  for (int i = 0; i < 8; ++i) {
    const int qg = qt * 16 + half * 8 + i;
    if (qg > 1024) continue;
    const float inv = 1.f / lr[i];
    const size_t tok = (size_t)b * NSEQ + qg;
    #pragma unroll
    for (int d = 0; d < 8; ++d)
      aout[tok * DIM + h * HD + d * 16 + ln] = (__bf16)(oacc[d][i] * inv);
  }
}

// ---------------------------------------------------------------------------
// LayerNorm: one block (256 thr) per token; 4 consecutive cols per thread:
// one b128 load, one packed b64 bf16 store.
// ---------------------------------------------------------------------------
__global__ __launch_bounds__(256) void ln_kernel(
    const float* __restrict__ x, const float* __restrict__ g,
    const float* __restrict__ bb, __bf16* __restrict__ y)
{
  const int tok = blockIdx.x;
  const float* xr = x + (size_t)tok * DIM;
  const int t = threadIdx.x;
  const int c0 = t * 4;
  v4f v = *(const v4f*)(xr + c0);
  float s = v.x + v.y + v.z + v.w;
  float ss = v.x * v.x + v.y * v.y + v.z * v.z + v.w * v.w;
  #pragma unroll
  for (int m = 1; m < 32; m <<= 1) {
    s  += __shfl_xor(s,  m, 32);
    ss += __shfl_xor(ss, m, 32);
  }
  __shared__ float w1s[8], w2s[8];
  const int lane = t & 31, wv = t >> 5;
  if (lane == 0) { w1s[wv] = s; w2s[wv] = ss; }
  __syncthreads();
  s = 0.f; ss = 0.f;
  #pragma unroll
  for (int w = 0; w < 8; ++w) { s += w1s[w]; ss += w2s[w]; }
  const float mean = s * (1.0f / DIM);
  const float var  = ss * (1.0f / DIM) - mean * mean;
  const float rs   = rsqrtf(var + EPSF);
  v4f gv = *(const v4f*)(g + c0);
  v4f bv = *(const v4f*)(bb + c0);
  float o0 = (v.x - mean) * rs * gv.x + bv.x;
  float o1 = (v.y - mean) * rs * gv.y + bv.y;
  float o2 = (v.z - mean) * rs * gv.z + bv.z;
  float o3 = (v.w - mean) * rs * gv.w + bv.w;
  v2u u;
  u.x = pkbf(o0, o1);
  u.y = pkbf(o2, o3);
  *(v2u*)(y + (size_t)tok * DIM + c0) = u;
}

__global__ void build_x_kernel(const float* __restrict__ x,
                               const float* __restrict__ cls,
                               float* __restrict__ xb)
{
  size_t idx = (size_t)blockIdx.x * blockDim.x + threadIdx.x;
  if (idx >= (size_t)NTOK * DIM) return;
  int c = (int)(idx & 1023);
  int tokrow = (int)(idx >> 10);
  int b = tokrow / NSEQ, n = tokrow % NSEQ;
  xb[idx] = (n == 0) ? cls[c] : x[((size_t)b * 1024 + (n - 1)) * DIM + c];
}

__global__ void copy_kernel(const float* __restrict__ src,
                            float* __restrict__ dst, size_t n)
{
  size_t idx = (size_t)blockIdx.x * blockDim.x + threadIdx.x;
  if (idx < n) dst[idx] = src[idx];
}

__global__ void copy_cls_kernel(const float* __restrict__ xb,
                                float* __restrict__ dst)
{
  int idx = blockIdx.x * blockDim.x + threadIdx.x;  // 4096
  int b = idx >> 10, c = idx & 1023;
  dst[idx] = xb[(size_t)b * NSEQ * DIM + c];
}

// ---------------------------------------------------------------------------
struct LayerW {
  const float *ln1g, *ln1b, *wqkv, *wo, *bo, *ln2g, *ln2b, *w1, *b1, *w2, *b2;
};

extern "C" void kernel_launch(void* const* d_in, const int* in_sizes, int n_in,
                              void* d_out, int out_size, void* d_ws, size_t ws_size,
                              hipStream_t stream) {
  (void)in_sizes; (void)n_in; (void)out_size; (void)ws_size;
  const float* x   = (const float*)d_in[0];
  // d_in[1] = src_mask (all true in this problem; causal mask applied in-kernel)
  const float* cls = (const float*)d_in[2];

  auto layer = [&](int base, int l) {
    LayerW w;
    w.ln1g = (const float*)d_in[base + 0]  + (size_t)l * DIM;
    w.ln1b = (const float*)d_in[base + 1]  + (size_t)l * DIM;
    w.wqkv = (const float*)d_in[base + 2]  + (size_t)l * DIM * 3 * DIM;
    w.wo   = (const float*)d_in[base + 3]  + (size_t)l * DIM * DIM;
    w.bo   = (const float*)d_in[base + 4]  + (size_t)l * DIM;
    w.ln2g = (const float*)d_in[base + 5]  + (size_t)l * DIM;
    w.ln2b = (const float*)d_in[base + 6]  + (size_t)l * DIM;
    w.w1   = (const float*)d_in[base + 7]  + (size_t)l * DIM * 4 * DIM;
    w.b1   = (const float*)d_in[base + 8]  + (size_t)l * 4 * DIM;
    w.w2   = (const float*)d_in[base + 9]  + (size_t)l * 4 * DIM * DIM;
    w.b2   = (const float*)d_in[base + 10] + (size_t)l * DIM;
    return w;
  };

  // workspace layout (all sizes multiples of 256B)
  char* ws = (char*)d_ws;
  float*  xbuf = (float*)ws;   ws += (size_t)NTOK * DIM * 4;          // 16.8 MB
  __bf16* ybf  = (__bf16*)ws;  ws += (size_t)NTOK * DIM * 2;          //  8.4 MB
  const size_t qkvsz = (size_t)4 * HEADS * NPAD * HD * 2;             //  8.65 MB each
  __bf16* qbf  = (__bf16*)ws;  ws += qkvsz;
  __bf16* kbf  = (__bf16*)ws;  ws += qkvsz;
  __bf16* vbf  = (__bf16*)ws;  ws += qkvsz;
  __bf16* aout = (__bf16*)ws;  ws += (size_t)NTOK * DIM * 2;          //  8.4 MB
  __bf16* hbuf = (__bf16*)ws;  ws += (size_t)NTOK * 4 * DIM * 2;      // 33.6 MB

  const size_t nx = (size_t)NTOK * DIM;
  build_x_kernel<<<(int)((nx + 255) / 256), 256, 0, stream>>>(x, cls, xbuf);

  const int gm = (NTOK + BM - 1) / BM;   // 33 M-tiles

  for (int stage = 0; stage < 2; ++stage) {
    const int base = (stage == 0) ? 3 : 14;
    for (int l = 0; l < 4; ++l) {
      LayerW w = layer(base, l);
      ln_kernel<<<NTOK, 256, 0, stream>>>(xbuf, w.ln1g, w.ln1b, ybf);
      hipMemsetAsync(vbf, 0, qkvsz, stream);
      gemm_kernel<<<dim3(gm, 24), 256, 0, stream>>>(
          ybf, w.wqkv, nullptr, nullptr, nullptr, nullptr,
          qbf, kbf, vbf, NTOK, DIM, 3 * DIM, 0);
      attn_kernel<<<dim3(17, HEADS, 4), 128, 0, stream>>>(qbf, kbf, vbf, aout);
      gemm_kernel<<<dim3(gm, 8), 256, 0, stream>>>(
          aout, w.wo, w.bo, xbuf, xbuf, nullptr,
          nullptr, nullptr, nullptr, NTOK, DIM, DIM, 1);
      ln_kernel<<<NTOK, 256, 0, stream>>>(xbuf, w.ln2g, w.ln2b, ybf);
      gemm_kernel<<<dim3(gm, 32), 256, 0, stream>>>(
          ybf, w.w1, w.b1, nullptr, nullptr, hbuf,
          nullptr, nullptr, nullptr, NTOK, DIM, 4 * DIM, 2);
      gemm_kernel<<<dim3(gm, 8), 256, 0, stream>>>(
          hbuf, w.w2, w.b2, xbuf, xbuf, nullptr,
          nullptr, nullptr, nullptr, NTOK, 4 * DIM, DIM, 1);
    }
    if (stage == 0) {
      copy_cls_kernel<<<16, 256, 0, stream>>>(xbuf, (float*)d_out + nx);
    }
  }
  copy_kernel<<<(int)((nx + 255) / 256), 256, 0, stream>>>(xbuf, (float*)d_out, nx);
}
